// SpatialSelfAttention_24885040513077
// MI455X (gfx1250) — matrix operand order repaired and checked by reading, not compiled, not run on hardware
//
#include <hip/hip_runtime.h>
#include <hip/hip_bf16.h>

// ---------------------------------------------------------------------------
// SpatialSelfAttention on MI455X (gfx1250): GroupNorm -> QKV -> flash attn ->
// out-proj + residual.  All GEMMs use v_wmma_f32_16x16x32_f16 (wave32).
// ---------------------------------------------------------------------------

typedef _Float16 v16h __attribute__((ext_vector_type(16)));
typedef _Float16 v8h  __attribute__((ext_vector_type(8)));
typedef float    v8f  __attribute__((ext_vector_type(8)));

#define C_DIM 512
#define HW    4096
#define B_DIM 4
#define NGRP  32
#define CPG   16   // channels per group

// ---- WMMA fragment helpers (layouts per CDNA5 ISA §7.12.2) -----

// A matrix 16x32 f16, source row-major [row, K] with leading dim `ld`.
// lane m=lane&15 -> row; half=lane>>4; elem e -> K = k0+half*16+e
// (contiguous, matches load_b_nmajor).
static __device__ __forceinline__ v16h load_a_row(const _Float16* __restrict__ src,
                                                  int ld, int row0, int k0) {
  const int lane = threadIdx.x & 31;
  const int m  = lane & 15;
  const int hh = lane >> 4;
  const _Float16* p = src + (size_t)(row0 + m) * ld + k0 + hh * 16;
  v8h lo = *(const v8h*)(p);
  v8h hi = *(const v8h*)(p + 8);
  v16h a;
#pragma unroll
  for (int e = 0; e < 8; ++e) { a[e] = lo[e]; a[e + 8] = hi[e]; }
  return a;
}

// B matrix 32x16 f16 from N-major storage: B[K][N] = src[(n0+N)*ld + k0 + K'].
// lane n=lane&15 -> column; half=lane>>4; elem e -> K = half*16 + e.
// One contiguous 32-byte load per lane.
static __device__ __forceinline__ v16h load_b_nmajor(const _Float16* __restrict__ src,
                                                     int ld, int n0, int k0) {
  const int lane = threadIdx.x & 31;
  const int n  = lane & 15;
  const int hh = lane >> 4;
  const _Float16* p = src + (size_t)(n0 + n) * ld + k0 + hh * 16;
  return *(const v16h*)(p);
}

static __device__ __forceinline__ v8f wmma_f16(v16h a, v16h b, v8f c) {
  // (neg_a, A, neg_b, B, c_mod, C, reuse_a, reuse_b)
  return __builtin_amdgcn_wmma_f32_16x16x32_f16(false, a, false, b, (short)0, c,
                                                false, false);
}

// ---------------------------------------------------------------------------
// Kernel 1: GroupNorm(x) -> hT f16, token-major [b][hw][c]
// One block per (group, batch): 16 channels x 4096 pixels.
// ---------------------------------------------------------------------------
__global__ __launch_bounds__(256) void gn_kernel(const float* __restrict__ x,
                                                 const float* __restrict__ gamma,
                                                 const float* __restrict__ beta,
                                                 _Float16* __restrict__ hT) {
  const int g = blockIdx.x;   // 0..31
  const int b = blockIdx.y;   // 0..3
  const int tid = threadIdx.x;
  const int N = CPG * HW;     // 65536 elements per group
  const float* xb = x + ((size_t)b * C_DIM + g * CPG) * HW;

  float s = 0.f, s2 = 0.f;
  for (int idx = tid; idx < N; idx += 256) {
    float v = xb[idx];
    s += v; s2 += v * v;
  }
  __shared__ float red0[256], red1[256];
  red0[tid] = s; red1[tid] = s2;
  __syncthreads();
#pragma unroll
  for (int off = 128; off > 0; off >>= 1) {
    if (tid < off) { red0[tid] += red0[tid + off]; red1[tid] += red1[tid + off]; }
    __syncthreads();
  }
  const float invN = 1.0f / (float)N;
  const float mean = red0[0] * invN;
  const float var  = red1[0] * invN - mean * mean;
  const float inv  = rsqrtf(var + 1e-6f);

  _Float16* hb = hT + (size_t)b * HW * C_DIM;
  for (int idx = tid; idx < N; idx += 256) {
    int cl = idx >> 12;          // channel within group
    int i  = idx & (HW - 1);     // pixel
    int c  = g * CPG + cl;
    float v = (xb[idx] - mean) * inv * gamma[c] + beta[c];
    hb[(size_t)i * C_DIM + c] = (_Float16)v;
  }
}

// ---------------------------------------------------------------------------
// Kernel 2: fp32 -> f16 weight conversion
// ---------------------------------------------------------------------------
__global__ void f2h_kernel(const float* __restrict__ src,
                           _Float16* __restrict__ dst, int n) {
  int i = blockIdx.x * blockDim.x + threadIdx.x;
  if (i < n) dst[i] = (_Float16)src[i];
}

// ---------------------------------------------------------------------------
// Kernel 3: QKV projections.  q/k stored token-major qT/kT [hw][c];
// v stored channel-major [c][hw].
// Each wave computes a 16(o) x 64(i) strip: one A (weight) fragment per
// k-step reused across 4 wmma; the 4 B fragments are shared by all 8 waves
// of the block through L0.  grid: (HW/64, 4 o-groups, batch*3), block 256.
// ---------------------------------------------------------------------------
__global__ __launch_bounds__(256) void qkv_kernel(
    const _Float16* __restrict__ hT,
    const _Float16* __restrict__ wq16, const _Float16* __restrict__ wk16,
    const _Float16* __restrict__ wv16,
    const float* __restrict__ bq, const float* __restrict__ bk,
    const float* __restrict__ bv,
    _Float16* __restrict__ qT, _Float16* __restrict__ kT,
    _Float16* __restrict__ vC) {
  const int z = blockIdx.z;
  const int b = z / 3, mat = z % 3;
  const _Float16* w    = (mat == 0) ? wq16 : (mat == 1) ? wk16 : wv16;
  const float*    bias = (mat == 0) ? bq   : (mat == 1) ? bk   : bv;
  const _Float16* hTb  = hT + (size_t)b * HW * C_DIM;

  const int wave = threadIdx.x >> 5;
  const int lane = threadIdx.x & 31;
  const int hh = lane >> 4, n = lane & 15;
  const int i0 = blockIdx.x * 64;
  const int o0 = (blockIdx.y * 8 + wave) * 16;

  v8f acc[4];
#pragma unroll
  for (int t = 0; t < 4; ++t)
#pragma unroll
    for (int e = 0; e < 8; ++e) acc[t][e] = 0.f;

#pragma unroll 2
  for (int kk = 0; kk < C_DIM; kk += 32) {
    v16h a = load_a_row(w, C_DIM, o0, kk);          // W[o, c] row-major
#pragma unroll
    for (int t = 0; t < 4; ++t) {
      v16h bm = load_b_nmajor(hTb, C_DIM, i0 + t * 16, kk);  // h[c,i]=hT[i*512+c]
      acc[t] = wmma_f16(a, bm, acc[t]);
    }
  }

  if (mat == 2) {
    _Float16* vb = vC + (size_t)b * C_DIM * HW;
#pragma unroll
    for (int t = 0; t < 4; ++t)
#pragma unroll
      for (int r = 0; r < 8; ++r) {
        int o = o0 + hh * 8 + r;
        vb[(size_t)o * HW + i0 + t * 16 + n] = (_Float16)(acc[t][r] + bias[o]);
      }
  } else {
    _Float16* ob = ((mat == 0) ? qT : kT) + (size_t)b * HW * C_DIM;
#pragma unroll
    for (int t = 0; t < 4; ++t)
#pragma unroll
      for (int r = 0; r < 8; ++r) {
        int o = o0 + hh * 8 + r;
        ob[(size_t)(i0 + t * 16 + n) * C_DIM + o] = (_Float16)(acc[t][r] + bias[o]);
      }
  }
}

// ---------------------------------------------------------------------------
// Kernel 4: flash attention.  Block = 512 threads (16 waves) owns 64 queries.
// Streams 64-key tiles: S phase (16 waves x one 16x16 S tile, full K=512),
// online softmax in LDS (4 threads per row), O phase (waves = q-tile x
// 128-channel block, P A-fragments loaded once per k-step from LDS).
// grid: (HW/64, batch).
// ---------------------------------------------------------------------------
__global__ __launch_bounds__(512, 1) void attn_kernel(
    const _Float16* __restrict__ qT, const _Float16* __restrict__ kT,
    const _Float16* __restrict__ vC, _Float16* __restrict__ oT) {
  const int b  = blockIdx.y;
  const int q0 = blockIdx.x * 64;
  const int tid  = threadIdx.x;
  const int wave = tid >> 5;
  const int lane = tid & 31;
  const int hh = lane >> 4;
  const int qt  = wave >> 2;   // query sub-tile 0..3
  const int sub = wave & 3;    // S phase: key sub-tile; O phase: channel block

  const _Float16* qTb = qT + (size_t)b * HW * C_DIM;
  const _Float16* kTb = kT + (size_t)b * HW * C_DIM;
  const _Float16* vCb = vC + (size_t)b * C_DIM * HW;
  _Float16*       oTb = oT + (size_t)b * HW * C_DIM;

  __shared__ float Sl[64 * 65];                       // pad 65 -> conflict-free rows
  __shared__ __align__(16) _Float16 Pl[64 * 72];      // ld=72 keeps 16B-aligned frags
  __shared__ float mrow[64], lrow[64], frow[64];
  __shared__ float pred[64 * 4];                      // partial max / sum

  if (tid < 64) { mrow[tid] = -3.0e38f; lrow[tid] = 0.0f; }

  v8f acc[8];
#pragma unroll
  for (int t = 0; t < 8; ++t)
#pragma unroll
    for (int e = 0; e < 8; ++e) acc[t][e] = 0.f;

  const float sscale = 0.044194173824159216f;  // 512^-0.5

  for (int jb = 0; jb < HW; jb += 64) {
    // ---- Phase 1: S tile (this wave: rows qt*16.., cols jb+sub*16..) ----
    v8f s;
#pragma unroll
    for (int e = 0; e < 8; ++e) s[e] = 0.f;
#pragma unroll 4
    for (int kk = 0; kk < C_DIM; kk += 32) {
      v16h a  = load_a_row(qTb, C_DIM, q0 + qt * 16, kk);
      v16h bm = load_b_nmajor(kTb, C_DIM, jb + sub * 16, kk);
      s = wmma_f16(a, bm, s);
    }
#pragma unroll
    for (int r = 0; r < 8; ++r)
      Sl[(qt * 16 + hh * 8 + r) * 65 + sub * 16 + (lane & 15)] = s[r] * sscale;
    __syncthreads();

    // ---- Phase 2: online softmax stats (4 threads per row) ----
    if (tid < 256) {                       // stage A: partial row max
      int row = tid >> 2, seg = tid & 3;
      const float* srp = &Sl[row * 65 + seg * 16];
      float mx = -3.0e38f;
#pragma unroll
      for (int j = 0; j < 16; ++j) mx = fmaxf(mx, srp[j]);
      pred[row * 4 + seg] = mx;
    }
    __syncthreads();
    if (tid < 64) {                        // stage B: combine, update m
      float mx = fmaxf(fmaxf(pred[tid * 4 + 0], pred[tid * 4 + 1]),
                       fmaxf(pred[tid * 4 + 2], pred[tid * 4 + 3]));
      mx = fmaxf(mx, mrow[tid]);
      frow[tid] = __expf(mrow[tid] - mx);  // 0 on first tile (m = -inf)
      mrow[tid] = mx;
    }
    __syncthreads();
    if (tid < 256) {                       // stage C: P = exp(S - m), partial sum
      int row = tid >> 2, seg = tid & 3;
      float mx = mrow[row];
      const float* srp = &Sl[row * 65 + seg * 16];
      _Float16* prp = &Pl[row * 72 + seg * 16];
      float sum = 0.f;
#pragma unroll
      for (int j = 0; j < 16; ++j) {
        float p = __expf(srp[j] - mx);
        prp[j] = (_Float16)p;
        sum += p;
      }
      pred[row * 4 + seg] = sum;
    }
    __syncthreads();
    if (tid < 64) {                        // stage D: combine sums, update l
      float sum = (pred[tid * 4 + 0] + pred[tid * 4 + 1]) +
                  (pred[tid * 4 + 2] + pred[tid * 4 + 3]);
      lrow[tid] = lrow[tid] * frow[tid] + sum;
    }
    __syncthreads();

    // ---- Phase 3: rescale O, accumulate P x V^T for 128 channels ----
    float f[8];
#pragma unroll
    for (int r = 0; r < 8; ++r) f[r] = frow[qt * 16 + hh * 8 + r];
#pragma unroll
    for (int t = 0; t < 8; ++t)
#pragma unroll
      for (int r = 0; r < 8; ++r) acc[t][r] *= f[r];

#pragma unroll
    for (int kk = 0; kk < 2; ++kk) {
      v16h a = load_a_row(Pl, 72, qt * 16, kk * 32);   // P from LDS, reused 8x
#pragma unroll
      for (int nt = 0; nt < 8; ++nt) {
        v16h bm = load_b_nmajor(vCb + jb, HW, sub * 128 + nt * 16, kk * 32);
        acc[nt] = wmma_f16(a, bm, acc[nt]);
      }
    }
    __syncthreads();
  }

  // ---- Finalize: divide by l, store O^T token-major f16 ----
  float il[8];
#pragma unroll
  for (int r = 0; r < 8; ++r) il[r] = 1.0f / lrow[qt * 16 + hh * 8 + r];
#pragma unroll
  for (int nt = 0; nt < 8; ++nt)
#pragma unroll
    for (int r = 0; r < 8; ++r) {
      int row = q0 + qt * 16 + hh * 8 + r;
      int col = sub * 128 + nt * 16 + (lane & 15);
      oTb[(size_t)row * C_DIM + col] = (_Float16)(acc[nt][r] * il[r]);
    }
}

// ---------------------------------------------------------------------------
// Kernel 5: out = x + Wo * O + bo  (fp32 output)
// Each wave: 16(o) x 64(i) strip, A fragment reused across 4 wmma.
// grid: (HW/64, 4 o-groups, batch), block 256 (8 waves).
// ---------------------------------------------------------------------------
__global__ __launch_bounds__(256) void proj_kernel(
    const float* __restrict__ x, const _Float16* __restrict__ oT,
    const _Float16* __restrict__ wo16, const float* __restrict__ bo,
    float* __restrict__ out) {
  const int b = blockIdx.z;
  const int wave = threadIdx.x >> 5;
  const int lane = threadIdx.x & 31;
  const int hh = lane >> 4, n = lane & 15;
  const int i0 = blockIdx.x * 64;
  const int o0 = (blockIdx.y * 8 + wave) * 16;
  const _Float16* oTb = oT + (size_t)b * HW * C_DIM;

  v8f acc[4];
#pragma unroll
  for (int t = 0; t < 4; ++t)
#pragma unroll
    for (int e = 0; e < 8; ++e) acc[t][e] = 0.f;

#pragma unroll 2
  for (int kk = 0; kk < C_DIM; kk += 32) {
    v16h a = load_a_row(wo16, C_DIM, o0, kk);
#pragma unroll
    for (int t = 0; t < 4; ++t) {
      v16h bm = load_b_nmajor(oTb, C_DIM, i0 + t * 16, kk);
      acc[t] = wmma_f16(a, bm, acc[t]);
    }
  }

#pragma unroll
  for (int t = 0; t < 4; ++t)
#pragma unroll
    for (int r = 0; r < 8; ++r) {
      int o = o0 + hh * 8 + r;
      size_t idx = ((size_t)b * C_DIM + o) * HW + i0 + t * 16 + n;
      out[idx] = x[idx] + bo[o] + acc[t][r];
    }
}

// ---------------------------------------------------------------------------
// Host launcher
// ---------------------------------------------------------------------------
extern "C" void kernel_launch(void* const* d_in, const int* in_sizes, int n_in,
                              void* d_out, int out_size, void* d_ws, size_t ws_size,
                              hipStream_t stream) {
  const float* x    = (const float*)d_in[0];
  const float* gn_w = (const float*)d_in[1];
  const float* gn_b = (const float*)d_in[2];
  const float* wq   = (const float*)d_in[3];
  const float* bq   = (const float*)d_in[4];
  const float* wk   = (const float*)d_in[5];
  const float* bk   = (const float*)d_in[6];
  const float* wv   = (const float*)d_in[7];
  const float* bv   = (const float*)d_in[8];
  const float* wo   = (const float*)d_in[9];
  const float* bo   = (const float*)d_in[10];
  float* out = (float*)d_out;

  // Workspace layout (f16), all offsets 256B-aligned.
  const size_t WSZ = (size_t)C_DIM * C_DIM;        // 262144 elems
  const size_t TSZ = (size_t)B_DIM * HW * C_DIM;   // 8388608 elems
  char* p = (char*)d_ws;
  _Float16* wq16 = (_Float16*)p; p += WSZ * 2;
  _Float16* wk16 = (_Float16*)p; p += WSZ * 2;
  _Float16* wv16 = (_Float16*)p; p += WSZ * 2;
  _Float16* wo16 = (_Float16*)p; p += WSZ * 2;
  _Float16* hT   = (_Float16*)p; p += TSZ * 2;
  _Float16* qT   = (_Float16*)p; p += TSZ * 2;
  _Float16* kT   = (_Float16*)p; p += TSZ * 2;
  _Float16* vC   = (_Float16*)p; p += TSZ * 2;
  _Float16* oT   = (_Float16*)p; p += TSZ * 2;

  // 1) weight conversion
  {
    int n = (int)WSZ, blk = 256, grd = (n + blk - 1) / blk;
    f2h_kernel<<<grd, blk, 0, stream>>>(wq, wq16, n);
    f2h_kernel<<<grd, blk, 0, stream>>>(wk, wk16, n);
    f2h_kernel<<<grd, blk, 0, stream>>>(wv, wv16, n);
    f2h_kernel<<<grd, blk, 0, stream>>>(wo, wo16, n);
  }
  // 2) GroupNorm -> hT
  gn_kernel<<<dim3(NGRP, B_DIM), 256, 0, stream>>>(x, gn_w, gn_b, hT);
  // 3) QKV
  qkv_kernel<<<dim3(HW / 64, 4, B_DIM * 3), 256, 0, stream>>>(
      hT, wq16, wk16, wv16, bq, bk, bv, qT, kT, vC);
  // 4) flash attention
  attn_kernel<<<dim3(HW / 64, B_DIM), 512, 0, stream>>>(qT, kT, vC, oT);
  // 5) projection + residual
  proj_kernel<<<dim3(HW / 64, 4, B_DIM), 256, 0, stream>>>(x, oT, wo16, bo, out);
}
